// LDPCEncoder_42064909697028
// MI455X (gfx1250) — 8-bit matrix operand order repaired and checked by reading, compiled, not run on hardware
//
#include <hip/hip_runtime.h>

typedef int   v16i __attribute__((ext_vector_type(16)));
typedef float v8f  __attribute__((ext_vector_type(8)));

#define BATCH   16384
#define NBITS   2400
#define KDIM    1200
#define KPAD    1280
#define TILES_N (NBITS / 16)
#define TILES_B (BATCH / 16)
#define FP8_ONE 0x38u

__global__ __launch_bounds__(256) void ldpc_conv_m(const int* __restrict__ m,
                                                   unsigned char* __restrict__ A8) {
    const long t = (long)blockIdx.x * blockDim.x + threadIdx.x;
    const int  b  = (int)(t / (KPAD / 4));
    const int  kq = (int)(t % (KPAD / 4)) * 4;
    unsigned int packed = 0u;
    if (kq < KDIM) {
        const int4 mv = *(const int4*)(m + (long)b * KDIM + kq);
        packed  = (mv.x ? FP8_ONE : 0u);
        packed |= (mv.y ? FP8_ONE : 0u) << 8;
        packed |= (mv.z ? FP8_ONE : 0u) << 16;
        packed |= (mv.w ? FP8_ONE : 0u) << 24;
    }
    *(unsigned int*)(A8 + (long)b * KPAD + kq) = packed;
}

__global__ __launch_bounds__(256) void ldpc_conv_G(const float* __restrict__ G,
                                                   unsigned char* __restrict__ G8) {
    const long t = (long)blockIdx.x * blockDim.x + threadIdx.x;
    const int  n  = (int)(t / (KPAD / 4));
    const int  kq = (int)(t % (KPAD / 4)) * 4;
    unsigned int packed = 0u;
    if (kq < KDIM) {
        const float4 gv = *(const float4*)(G + (long)n * KDIM + kq);
        packed  = (gv.x != 0.0f ? FP8_ONE : 0u);
        packed |= (gv.y != 0.0f ? FP8_ONE : 0u) << 8;
        packed |= (gv.z != 0.0f ? FP8_ONE : 0u) << 16;
        packed |= (gv.w != 0.0f ? FP8_ONE : 0u) << 24;
    }
    *(unsigned int*)(G8 + (long)n * KPAD + kq) = packed;
}

__global__ __launch_bounds__(256) void ldpc_wmma_fp8(const unsigned char* __restrict__ A8,
                                                     const unsigned char* __restrict__ B8,
                                                     float* __restrict__ out) {
    const int lane = threadIdx.x & 31;
    const int wave = threadIdx.x >> 5;
    const int tile = blockIdx.x * 8 + wave;
    const int tb   = tile / TILES_N;
    const int tn   = tile % TILES_N;
    const int r16  = lane & 15;
    const int half = lane >> 4;

    const unsigned char* __restrict__ arow = A8 + (long)(tb * 16 + r16) * KPAD;
    const unsigned char* __restrict__ brow = B8 + (long)(tn * 16 + r16) * KPAD;

    v8f c = {};
#pragma unroll
    for (int kc = 0; kc < KPAD; kc += 128) {
        v16i a, b;
#pragma unroll
        for (int p = 0; p < 4; ++p) {
            const int4 v = *(const int4*)(arow + kc + p * 32 + half * 16);
            a[4 * p] = v.x;  a[4 * p + 1] = v.y;  a[4 * p + 2] = v.z;  a[4 * p + 3] = v.w;
        }
#pragma unroll
        for (int q = 0; q < 4; ++q) {
            const int4 v = *(const int4*)(brow + kc + q * 32 + half * 16);
            b[4 * q] = v.x;  b[4 * q + 1] = v.y;  b[4 * q + 2] = v.z;  b[4 * q + 3] = v.w;
        }
        c = __builtin_amdgcn_wmma_f32_16x16x128_fp8_fp8(a, b, (short)0, c, false, false);
    }

    const int n = tn * 16 + r16;
#pragma unroll
    for (int j = 0; j < 8; ++j) {
        const int  mrow = tb * 16 + half * 8 + j;
        const int  bit  = ((int)c[j]) & 1;
        out[(long)mrow * NBITS + n] = 1.0f - 2.0f * (float)bit;
    }
}

extern "C" void kernel_launch(void* const* d_in, const int* in_sizes, int n_in,
                              void* d_out, int out_size, void* d_ws, size_t ws_size,
                              hipStream_t stream) {
    (void)in_sizes; (void)n_in; (void)out_size; (void)ws_size;
    const int*   m = (const int*)d_in[0];
    const float* G = (const float*)d_in[1];
    float*     out = (float*)d_out;

    unsigned char* A8 = (unsigned char*)d_ws;
    unsigned char* G8 = A8 + (size_t)BATCH * KPAD;

    ldpc_conv_m<<<(BATCH * (KPAD / 4)) / 256, 256, 0, stream>>>(m, A8);
    ldpc_conv_G<<<(NBITS * (KPAD / 4)) / 256, 256, 0, stream>>>(G, G8);
    ldpc_wmma_fp8<<<(TILES_B * TILES_N) / 8, 256, 0, stream>>>(A8, G8, out);
}
